// ImprovedMoE_74105365725742
// MI455X (gfx1250) — matrix operand order repaired and checked by reading, not compiled, not run on hardware
//
#include <hip/hip_runtime.h>
#include <cstdint>
#include <cstddef>

// ---------- CDNA5 WMMA types ----------
typedef __attribute__((ext_vector_type(16))) __bf16 v16bf;
typedef __attribute__((ext_vector_type(8)))  float  v8f;

typedef unsigned short us;

// ---------- bf16 helpers (storage = ushort bits) ----------
__device__ __forceinline__ float bf2f(us h) {
    union { unsigned int u; float f; } c; c.u = ((unsigned int)h) << 16; return c.f;
}
__device__ __forceinline__ us f2bf(float f) {
    union { float f; unsigned int u; } c; c.f = f;
    unsigned int u = c.u;
    unsigned int r = u + 0x7FFFu + ((u >> 16) & 1u);   // round-to-nearest-even
    return (us)(r >> 16);
}

union FragBF { v16bf v; uint4 q[2]; };
union Pack8  { us u[8]; uint4 q; };

__device__ __forceinline__ void asyncLdsB128(unsigned ldsOff, const void* gptr) {
    unsigned long long ga = (unsigned long long)(uintptr_t)gptr;
    asm volatile("global_load_async_to_lds_b128 %0, %1, off" :: "v"(ldsOff), "v"(ga) : "memory");
}
__device__ __forceinline__ void waitAsync() {
    asm volatile("s_wait_asynccnt 0" ::: "memory");
}

// =====================================================================
// Conversion / packing kernels
// =====================================================================
__global__ void k_cvt_bf16(const float* __restrict__ src, us* __restrict__ dst, int n) {
    int i = blockIdx.x * blockDim.x + threadIdx.x;
    if (i < n) dst[i] = f2bf(src[i]);
}

// x [N][3][H][W] f32 -> NHWC bf16 [N][H*W][3]
__global__ void k_cvt_x_nhwc(const float* __restrict__ x, us* __restrict__ dst, int n) {
    int i = blockIdx.x * blockDim.x + threadIdx.x;
    if (i >= n) return;
    int nn = i / (3 * 1024); int r = i % (3 * 1024); int c = r / 1024; int p = r % 1024;
    dst[((size_t)nn * 1024 + p) * 3 + c] = f2bf(x[i]);
}

// conv weights [Cout][Cin][3][3] f32 -> bf16 [Cout][Kp], k = r9*Cin + ci
__global__ void k_pack_convw(const float* __restrict__ w, us* __restrict__ dst,
                             int Cout, int Cin, int Kp) {
    int i = blockIdx.x * blockDim.x + threadIdx.x;
    if (i >= Cout * Kp) return;
    int co = i / Kp, k = i % Kp;
    int Kreal = Cin * 9;
    us v = 0;
    if (k < Kreal) {
        int r9 = k / Cin; int ci = k - r9 * Cin;
        v = f2bf(w[((size_t)co * Cin + ci) * 9 + r9]);
    }
    dst[i] = v;
}

// cls_w [16][10][128] f32 -> [16][16][128] bf16 (M padded to 16)
__global__ void k_pack_clsw(const float* __restrict__ w, us* __restrict__ dst) {
    int i = blockIdx.x * blockDim.x + threadIdx.x;
    if (i >= 16 * 16 * 128) return;
    int e = i / (16 * 128); int r = i % (16 * 128); int m = r / 128; int d = r % 128;
    dst[i] = (m < 10) ? f2bf(w[((size_t)e * 10 + m) * 128 + d]) : (us)0;
}

// =====================================================================
// Implicit-GEMM 3x3 SAME conv + BN + ReLU.  NHWC bf16, f32 acc.
// Block: 128 thr (4 waves). Block tile: 64 Cout x 64 px.
// Each wave: 2 A-frags x 2 B-frags -> 4 WMMAs per K-chunk.
// Double-buffered LDS, 1 barrier/chunk; A and B staged with
// global_load_async_to_lds_b128 (ASYNCcnt), next chunk overlaps WMMAs.
// =====================================================================
__global__ __launch_bounds__(128)
void k_conv3x3_wmma(const us* __restrict__ in,     // NHWC [N][H*W][Cin]
                    const us* __restrict__ wpack,  // [Cout][Kp] (r9-major K)
                    const float* __restrict__ convb,
                    const float* __restrict__ bns,
                    const float* __restrict__ bnb,
                    us* __restrict__ out,          // NHWC [N][H*W][Cout]
                    int Cin, int H, int W, int Cout, int Kp) {
    __shared__ __align__(16) us ldsA[2 * 64 * 32];   // [buf][cout][k]
    __shared__ __align__(16) us ldsB[2 * 64 * 32];   // [buf][px][k]
    const int t        = threadIdx.x;
    const int pxBase   = blockIdx.x * 64;
    const int coutBase = blockIdx.y * 64;
    const int HW   = H * W;
    const int lane = t & 31, wv = t >> 5;            // 4 waves
    const int half = lane >> 4, lr = lane & 15;
    const int tm0 = (wv >> 1) * 2;                   // cout subtiles {tm0, tm0+1}
    const int tn0 = (wv & 1) * 2;                    // px   subtiles {tn0, tn0+1}
    v8f acc00 = {0.f,0.f,0.f,0.f,0.f,0.f,0.f,0.f};
    v8f acc01 = acc00, acc10 = acc00, acc11 = acc00;

    // staging role: thread -> (row 0..63, 16-element column run {0,16})
    const int srow = t >> 1;
    const int scol = (t & 1) * 16;
    const int pxg  = pxBase + srow;
    const int n_i  = pxg / HW;
    const int rem  = pxg % HW;
    const int y_i  = rem / W;
    const int x_i  = rem % W;

    const unsigned ldsAoff = (unsigned)(uintptr_t)(&ldsA[srow * 32 + scol]);
    const unsigned ldsBoff = (unsigned)(uintptr_t)(&ldsB[srow * 32 + scol]);
    const us* wBase = wpack + (size_t)(coutBase + srow) * Kp + scol;

    auto stageFast = [&](int buf, int r9, int cb, int kIdx) {
        // A: weights, always in-bounds, 32B per thread = 2 async b128
        const us* wp = wBase + kIdx;
        asyncLdsB128(ldsAoff + buf * 4096,      wp);
        asyncLdsB128(ldsAoff + buf * 4096 + 16, wp + 8);
        // B: one (dy,dx) tap, 32 contiguous channels; halo lanes write zeros
        int dy = (r9 >= 6) ? 1 : ((r9 >= 3) ? 0 : -1);
        int dx = r9 - (dy + 1) * 3 - 1;
        int yy = y_i + dy, xx = x_i + dx;
        bool inb = (yy >= 0) && (yy < H) && (xx >= 0) && (xx < W);
        if (inb) {
            const us* ip = in + ((size_t)((size_t)n_i * H + yy) * W + xx) * Cin + cb + scol;
            asyncLdsB128(ldsBoff + buf * 4096,      ip);
            asyncLdsB128(ldsBoff + buf * 4096 + 16, ip + 8);
        } else {
            uint4 z = {0u, 0u, 0u, 0u};
            *reinterpret_cast<uint4*>(&ldsB[buf * 2048 + srow * 32 + scol])     = z;
            *reinterpret_cast<uint4*>(&ldsB[buf * 2048 + srow * 32 + scol + 8]) = z;
        }
    };

    auto compute = [&](int buf) {
        const us* A  = ldsA + buf * 2048;
        const us* Bt = ldsB + buf * 2048;
        FragBF fa0, fa1, fb0, fb1;
        const int a0 = (tm0 * 16 + lr) * 32, a1 = ((tm0 + 1) * 16 + lr) * 32;
        const int b0 = (tn0 * 16 + lr) * 32, b1 = ((tn0 + 1) * 16 + lr) * 32;
        fa0.q[0] = *reinterpret_cast<const uint4*>(A + a0 + 8 * half);
        fa0.q[1] = *reinterpret_cast<const uint4*>(A + a0 + 16 + 8 * half);
        fa1.q[0] = *reinterpret_cast<const uint4*>(A + a1 + 8 * half);
        fa1.q[1] = *reinterpret_cast<const uint4*>(A + a1 + 16 + 8 * half);
        fb0.q[0] = *reinterpret_cast<const uint4*>(Bt + b0 + 8 * half);
        fb0.q[1] = *reinterpret_cast<const uint4*>(Bt + b0 + 16 + 8 * half);
        fb1.q[0] = *reinterpret_cast<const uint4*>(Bt + b1 + 8 * half);
        fb1.q[1] = *reinterpret_cast<const uint4*>(Bt + b1 + 16 + 8 * half);
        acc00 = __builtin_amdgcn_wmma_f32_16x16x32_bf16(false, fa0.v, false, fb0.v, (short)0, acc00, false, false);
        acc01 = __builtin_amdgcn_wmma_f32_16x16x32_bf16(false, fa0.v, false, fb1.v, (short)0, acc01, false, false);
        acc10 = __builtin_amdgcn_wmma_f32_16x16x32_bf16(false, fa1.v, false, fb0.v, (short)0, acc10, false, false);
        acc11 = __builtin_amdgcn_wmma_f32_16x16x32_bf16(false, fa1.v, false, fb1.v, (short)0, acc11, false, false);
    };

    if ((Cin & 31) == 0) {
        const int total = 9 * (Cin >> 5);
        int nr9 = 0, ncb = 0, nk = 0;
        // prologue: stage chunk 0 into buf 0
        stageFast(0, nr9, ncb, nk);
        ncb += 32; nk += 32; if (ncb == Cin) { ncb = 0; ++nr9; }
        int cur = 0;
        for (int c = 0; c < total; ++c) {
            waitAsync();
            __syncthreads();
            if (c + 1 < total) {
                stageFast(cur ^ 1, nr9, ncb, nk);
                ncb += 32; nk += 32; if (ncb == Cin) { ncb = 0; ++nr9; }
            }
            compute(cur);
            cur ^= 1;
        }
    } else {
        // generic path (conv1, Cin=3): per-element gather, Kp/32 chunks
        const int Kreal = Cin * 9;
        for (int kk = 0; kk < Kp; kk += 32) {
            const us* wp = wBase + kk;
            asyncLdsB128(ldsAoff,      wp);
            asyncLdsB128(ldsAoff + 16, wp + 8);
            us tmp[16];
            #pragma unroll
            for (int j = 0; j < 16; ++j) {
                int K = kk + scol + j;
                us v = 0;
                if (K < Kreal) {
                    int r9 = K / Cin, ci = K - r9 * Cin;
                    int yy = y_i + r9 / 3 - 1, xx = x_i + r9 % 3 - 1;
                    if (yy >= 0 && yy < H && xx >= 0 && xx < W)
                        v = in[((size_t)((size_t)n_i * H + yy) * W + xx) * Cin + ci];
                }
                tmp[j] = v;
            }
            *reinterpret_cast<uint4*>(&ldsB[srow * 32 + scol])     = *reinterpret_cast<uint4*>(&tmp[0]);
            *reinterpret_cast<uint4*>(&ldsB[srow * 32 + scol + 8]) = *reinterpret_cast<uint4*>(&tmp[8]);
            waitAsync();
            __syncthreads();
            compute(0);
            __syncthreads();
        }
    }

    // epilogue: per lane, 8 consecutive Cout -> one packed 16B NHWC store per subtile
    auto storeT = [&](const v8f& a, int tmI, int tnI) {
        const int coB = coutBase + tmI * 16 + 8 * half;
        const int pg  = pxBase + tnI * 16 + lr;
        Pack8 pk;
        #pragma unroll
        for (int r = 0; r < 8; ++r) {
            int co = coB + r;
            float v = bns[co] * (a[r] + convb[co]) + bnb[co];
            pk.u[r] = f2bf(v > 0.f ? v : 0.f);
        }
        *reinterpret_cast<uint4*>(&out[(size_t)pg * Cout + coB]) = pk.q;
    };
    storeT(acc00, tm0,     tn0);
    storeT(acc01, tm0,     tn0 + 1);
    storeT(acc10, tm0 + 1, tn0);
    storeT(acc11, tm0 + 1, tn0 + 1);
}

// =====================================================================
// 2x2 max pool, NHWC bf16
// =====================================================================
__global__ void k_maxpool(const us* __restrict__ in, us* __restrict__ out,
                          int C, int H, int W, int total) {
    int i = blockIdx.x * blockDim.x + threadIdx.x;
    if (i >= total) return;
    int Ho = H / 2, Wo = W / 2;
    int c = i % C; int p = i / C;
    int xo = p % Wo; int t1 = p / Wo; int yo = t1 % Ho; int n = t1 / Ho;
    size_t b00 = ((size_t)((size_t)n * H + 2 * yo) * W + 2 * xo) * C + c;
    float a = bf2f(in[b00]),                 b = bf2f(in[b00 + C]);
    float d = bf2f(in[b00 + (size_t)W * C]), e = bf2f(in[b00 + (size_t)W * C + C]);
    out[i] = f2bf(fmaxf(fmaxf(a, b), fmaxf(d, e)));
}

// global average pool over HW, NHWC -> feats bf16 [N][256]
__global__ void k_avgpool(const us* __restrict__ in, us* __restrict__ feats, int C, int HW) {
    int n = blockIdx.x, c = threadIdx.x;
    float s = 0.f;
    for (int p = 0; p < HW; ++p) s += bf2f(in[((size_t)n * HW + p) * C + c]);
    feats[(size_t)n * C + c] = f2bf(s / (float)HW);
}

// =====================================================================
// Batched per-expert GEMM: out[b][e][o] = sum_i in * W[e][o][i] + bias[e][o]
// 256 thr (8 waves), each wave one linearized 16x16 tile. bf16 WMMA.
// =====================================================================
__global__ __launch_bounds__(256)
void k_gemm_moe_wmma(const us* __restrict__ in,   // shared [B][K] or per-exp [B][E][K]
                     const us* __restrict__ Wt,   // [E][Mp][K] bf16
                     const float* __restrict__ bias,
                     float* __restrict__ out,     // [B][E][O]
                     int K, int Mp, int O, int perExpert, int E,
                     int tilesM, int totalTiles) {
    const int wv = threadIdx.x >> 5, lane = threadIdx.x & 31;
    const int tile = blockIdx.x * 8 + wv;            // wave-uniform
    if (tile >= totalTiles) return;
    const int perE = 32 * tilesM;                    // B/16 = 32 tiles in batch dim
    const int e  = tile / perE;
    const int r  = tile % perE;
    const int nb = (r / tilesM) * 16;
    const int mb = (r % tilesM) * 16;
    const int half = lane >> 4, lr = lane & 15;
    v8f acc = {0.f,0.f,0.f,0.f,0.f,0.f,0.f,0.f};
    const us* wrow = Wt + ((size_t)e * Mp + mb + lr) * K;
    const us* irow = perExpert ? in + ((size_t)(nb + lr) * E + e) * K
                               : in + (size_t)(nb + lr) * K;
    for (int kk = 0; kk < K; kk += 32) {
        FragBF fa, fb;
        fa.q[0] = *reinterpret_cast<const uint4*>(wrow + kk + 8 * half);
        fa.q[1] = *reinterpret_cast<const uint4*>(wrow + kk + 16 + 8 * half);
        fb.q[0] = *reinterpret_cast<const uint4*>(irow + kk + 8 * half);
        fb.q[1] = *reinterpret_cast<const uint4*>(irow + kk + 16 + 8 * half);
        acc = __builtin_amdgcn_wmma_f32_16x16x32_bf16(false, fa.v, false, fb.v,
                                                      (short)0, acc, false, false);
    }
    const int b = nb + lr;
    #pragma unroll
    for (int r8 = 0; r8 < 8; ++r8) {
        int m = mb + r8 + 8 * half;
        if (m < O)
            out[((size_t)b * E + e) * O + m] = acc[r8] + bias[(size_t)e * O + m];
    }
}

// =====================================================================
// LayerNorm (+gamma,beta) + ReLU, f32 in -> bf16 out. One wave per row.
// =====================================================================
__global__ __launch_bounds__(32)
void k_ln_relu_bf(const float* __restrict__ x, const float* __restrict__ g,
                  const float* __restrict__ bb, us* __restrict__ out, int O, int E) {
    int row = blockIdx.x;       // b*E + e
    int e = row % E;
    int lane = threadIdx.x;
    const float* p = x + (size_t)row * O;
    float s = 0.f, sq = 0.f;
    for (int i = lane; i < O; i += 32) { float v = p[i]; s += v; sq += v * v; }
    for (int m = 16; m >= 1; m >>= 1) { s += __shfl_xor(s, m, 32); sq += __shfl_xor(sq, m, 32); }
    float mean = s / (float)O;
    float var  = sq / (float)O - mean * mean;
    float inv  = rsqrtf(var + 1e-5f);
    for (int i = lane; i < O; i += 32) {
        float v = (p[i] - mean) * inv * g[(size_t)e * O + i] + bb[(size_t)e * O + i];
        v = v > 0.f ? v : 0.f;
        out[(size_t)row * O + i] = f2bf(v);
    }
}

__global__ void k_relu_bf(const float* __restrict__ x, us* __restrict__ out, int n) {
    int i = blockIdx.x * blockDim.x + threadIdx.x;
    if (i < n) { float v = x[i]; out[i] = f2bf(v > 0.f ? v : 0.f); }
}
__global__ void k_tobf(const float* __restrict__ x, us* __restrict__ out, int n) {
    int i = blockIdx.x * blockDim.x + threadIdx.x;
    if (i < n) out[i] = f2bf(x[i]);
}

// softmax over 10 logits -> confidence = -entropy. One wave per row.
__global__ __launch_bounds__(32)
void k_conf(const float* __restrict__ logits, float* __restrict__ conf) {
    int row = blockIdx.x; int lane = threadIdx.x;
    float x = (lane < 10) ? logits[(size_t)row * 10 + lane] : -3.0e38f;
    float mx = x;
    for (int m = 16; m >= 1; m >>= 1) mx = fmaxf(mx, __shfl_xor(mx, m, 32));
    float p = (lane < 10) ? expf(x - mx) : 0.f;
    float s = p;
    for (int m = 16; m >= 1; m >>= 1) s += __shfl_xor(s, m, 32);
    float pr = p / s;
    float t  = (lane < 10) ? pr * logf(pr + 1e-12f) : 0.f;
    for (int m = 16; m >= 1; m >>= 1) t += __shfl_xor(t, m, 32);
    if (lane == 0) conf[row] = t;
}

// gate layer 3: scores[b][e] = dot(g2[b][e][:32], gate_w3[e][0][:32]) + gate_b3[e]
__global__ __launch_bounds__(32)
void k_gate3(const us* __restrict__ g2, const float* __restrict__ w3,
             const float* __restrict__ b3, float* __restrict__ scores, int E) {
    int row = blockIdx.x; int e = row % E; int lane = threadIdx.x;
    float v = bf2f(g2[(size_t)row * 32 + lane]) * w3[(size_t)e * 32 + lane];
    for (int m = 16; m >= 1; m >>= 1) v += __shfl_xor(v, m, 32);
    if (lane == 0) scores[row] = v + b3[e];   // RTEMP = 1
}

// balanced = 0.7*scores + 0.3*conf - 2.0*(1/16)   (min-use boost is identically 0)
__global__ void k_balanced(const float* __restrict__ scores, const float* __restrict__ conf,
                           float* __restrict__ bal_ws, float* __restrict__ bal_out, int n) {
    int i = blockIdx.x * blockDim.x + threadIdx.x;
    if (i < n) {
        float b = 0.7f * scores[i] + 0.3f * conf[i] - 2.0f * 0.0625f;
        bal_ws[i] = b; bal_out[i] = b;
    }
}

// sequential capacity routing (B=512, E=16, K=2, CAP=64) + final gather
__global__ void k_route(const float* __restrict__ bal, const float* __restrict__ logitsE,
                        float* __restrict__ outLogits, float* __restrict__ outD) {
    if (threadIdx.x != 0 || blockIdx.x != 0) return;
    float loads[16];
    for (int e = 0; e < 16; ++e) loads[e] = 0.f;
    for (int b = 0; b < 512; ++b) {
        const float* r = bal + b * 16;
        int i0 = 0; float v0 = r[0];
        for (int e = 1; e < 16; ++e) if (r[e] > v0) { v0 = r[e]; i0 = e; }   // first max
        int i1 = -1; float v1 = -3.0e38f;
        for (int e = 0; e < 16; ++e) { if (e == i0) continue; if (r[e] > v1) { v1 = r[e]; i1 = e; } }
        float l0 = loads[i0], l1 = loads[i1];
        int chosen;
        if (l0 < 64.f)      chosen = i0;
        else if (l1 < 64.f) chosen = i1;
        else                chosen = (l0 <= l1) ? i0 : i1;
        loads[chosen] += 1.f;
        for (int e = 0; e < 16; ++e) outD[b * 16 + e] = (e == chosen) ? 1.f : 0.f;
        const float* lg = logitsE + ((size_t)b * 16 + chosen) * 10;
        for (int c = 0; c < 10; ++c) outLogits[b * 10 + c] = lg[c];
    }
}

// =====================================================================
// Host orchestration
// =====================================================================
extern "C" void kernel_launch(void* const* d_in, const int* in_sizes, int n_in,
                              void* d_out, int out_size, void* d_ws, size_t ws_size,
                              hipStream_t stream) {
    (void)in_sizes; (void)n_in; (void)out_size; (void)ws_size;
    const int B = 512, E = 16;

    const float* x = (const float*)d_in[0];
    const float *conv_w[6], *conv_b[6], *bn_s[6], *bn_b[6];
    for (int l = 0; l < 6; ++l) {
        conv_w[l] = (const float*)d_in[1 + 4 * l];
        conv_b[l] = (const float*)d_in[2 + 4 * l];
        bn_s[l]   = (const float*)d_in[3 + 4 * l];
        bn_b[l]   = (const float*)d_in[4 + 4 * l];
    }
    const float* gate_w1 = (const float*)d_in[25]; const float* gate_b1 = (const float*)d_in[26];
    const float* gate_g1 = (const float*)d_in[27]; const float* gate_bb1 = (const float*)d_in[28];
    const float* gate_w2 = (const float*)d_in[29]; const float* gate_b2 = (const float*)d_in[30];
    const float* gate_w3 = (const float*)d_in[31]; const float* gate_b3 = (const float*)d_in[32];
    const float* exp_w1 = (const float*)d_in[33];  const float* exp_b1 = (const float*)d_in[34];
    const float* exp_g1 = (const float*)d_in[35];  const float* exp_bb1 = (const float*)d_in[36];
    const float* exp_w2 = (const float*)d_in[37];  const float* exp_b2 = (const float*)d_in[38];
    const float* exp_g2 = (const float*)d_in[39];  const float* exp_bb2 = (const float*)d_in[40];
    const float* exp_w3 = (const float*)d_in[41];  const float* exp_b3 = (const float*)d_in[42];
    const float* cls_w  = (const float*)d_in[43];  const float* cls_b  = (const float*)d_in[44];

    // ---- workspace layout ----
    char* ws = (char*)d_ws;
    size_t off = 0;
    auto alloc = [&](size_t bytes) -> char* {
        char* p = ws + off; off += (bytes + 255) & ~(size_t)255; return p;
    };
    us* xbf   = (us*)alloc((size_t)B * 3 * 1024 * 2);      // NHWC
    us* actA  = (us*)alloc((size_t)B * 64 * 1024 * 2);     // NHWC ping
    us* actB  = (us*)alloc((size_t)B * 64 * 1024 * 2);     // NHWC pong
    const int Ci[6] = {3, 64, 64, 128, 128, 256};
    const int Co[6] = {64, 64, 128, 128, 256, 256};
    const int Kp[6] = {32, 576, 576, 1152, 1152, 2304};    // 9*Cin for fast layers
    us* wpk[6];
    for (int l = 0; l < 6; ++l) wpk[l] = (us*)alloc((size_t)Co[l] * Kp[l] * 2);
    us* expw1bf = (us*)alloc((size_t)E * 256 * 256 * 2);
    us* expw2bf = (us*)alloc((size_t)E * 128 * 256 * 2);
    us* expw3bf = (us*)alloc((size_t)E * 128 * 128 * 2);
    us* gatw1bf = (us*)alloc((size_t)E * 64 * 256 * 2);
    us* gatw2bf = (us*)alloc((size_t)E * 32 * 64 * 2);
    us* clsPad  = (us*)alloc((size_t)E * 16 * 128 * 2);
    us* featsbf = (us*)alloc((size_t)B * 256 * 2);
    float* G    = (float*)alloc((size_t)B * E * 256 * 4);
    us* Hb1     = (us*)alloc((size_t)B * E * 256 * 2);
    us* Hb2     = (us*)alloc((size_t)B * E * 256 * 2);
    float* logitsE = (float*)alloc((size_t)B * E * 10 * 4);
    float* conf    = (float*)alloc((size_t)B * E * 4);
    float* scores  = (float*)alloc((size_t)B * E * 4);
    float* balWs   = (float*)alloc((size_t)B * E * 4);

    float* outLogits = (float*)d_out;
    float* outBal    = outLogits + B * 10;
    float* outD      = outBal + B * E;

    auto cvt = [&](const float* s, us* d, int n) {
        k_cvt_bf16<<<(n + 255) / 256, 256, 0, stream>>>(s, d, n);
    };

    // ---- weight / input conversion ----
    k_cvt_x_nhwc<<<(B * 3 * 1024 + 255) / 256, 256, 0, stream>>>(x, xbf, B * 3 * 1024);
    for (int l = 0; l < 6; ++l) {
        int n = Co[l] * Kp[l];
        k_pack_convw<<<(n + 255) / 256, 256, 0, stream>>>(conv_w[l], wpk[l], Co[l], Ci[l], Kp[l]);
    }
    cvt(exp_w1, expw1bf, E * 256 * 256);
    cvt(exp_w2, expw2bf, E * 128 * 256);
    cvt(exp_w3, expw3bf, E * 128 * 128);
    cvt(gate_w1, gatw1bf, E * 64 * 256);
    cvt(gate_w2, gatw2bf, E * 32 * 64);
    k_pack_clsw<<<(E * 16 * 128 + 255) / 256, 256, 0, stream>>>(cls_w, clsPad);

    // ---- conv trunk (NHWC) ----
    auto conv = [&](const us* in, int l, us* out, int H, int W) {
        int P = B * H * W;
        dim3 g(P / 64, Co[l] / 64);
        k_conv3x3_wmma<<<g, 128, 0, stream>>>(in, wpk[l], conv_b[l], bn_s[l], bn_b[l],
                                              out, Ci[l], H, W, Co[l], Kp[l]);
    };
    conv(xbf,  0, actA, 32, 32);
    conv(actA, 1, actB, 32, 32);
    { int tot = B * 64 * 16 * 16;
      k_maxpool<<<(tot + 255) / 256, 256, 0, stream>>>(actB, actA, 64, 32, 32, tot); }
    conv(actA, 2, actB, 16, 16);
    conv(actB, 3, actA, 16, 16);
    { int tot = B * 128 * 8 * 8;
      k_maxpool<<<(tot + 255) / 256, 256, 0, stream>>>(actA, actB, 128, 16, 16, tot); }
    conv(actB, 4, actA, 8, 8);
    conv(actA, 5, actB, 8, 8);
    k_avgpool<<<B, 256, 0, stream>>>(actB, featsbf, 256, 64);

    // ---- experts ----
    auto gemm = [&](const us* in, const us* W, const float* bias, float* out,
                    int K, int Mp, int O, int perExp) {
        int tilesM = Mp / 16;
        int totalTiles = E * (B / 16) * tilesM;
        k_gemm_moe_wmma<<<(totalTiles + 7) / 8, 256, 0, stream>>>(
            in, W, bias, out, K, Mp, O, perExp, E, tilesM, totalTiles);
    };
    gemm(featsbf, expw1bf, exp_b1, G, 256, 256, 256, 0);
    k_ln_relu_bf<<<B * E, 32, 0, stream>>>(G, exp_g1, exp_bb1, Hb1, 256, E);
    gemm(Hb1, expw2bf, exp_b2, G, 256, 128, 128, 1);
    k_ln_relu_bf<<<B * E, 32, 0, stream>>>(G, exp_g2, exp_bb2, Hb2, 128, E);
    gemm(Hb2, expw3bf, exp_b3, G, 128, 128, 128, 1);          // embeds f32
    k_tobf<<<(B * E * 128 + 255) / 256, 256, 0, stream>>>(G, Hb1, B * E * 128);
    gemm(Hb1, clsPad, cls_b, logitsE, 128, 16, 10, 1);        // logits_e [B][E][10]
    k_conf<<<B * E, 32, 0, stream>>>(logitsE, conf);

    // ---- gates ----
    gemm(featsbf, gatw1bf, gate_b1, G, 256, 64, 64, 0);
    k_ln_relu_bf<<<B * E, 32, 0, stream>>>(G, gate_g1, gate_bb1, Hb2, 64, E);
    gemm(Hb2, gatw2bf, gate_b2, G, 64, 32, 32, 1);
    k_relu_bf<<<(B * E * 32 + 255) / 256, 256, 0, stream>>>(G, Hb1, B * E * 32);
    k_gate3<<<B * E, 32, 0, stream>>>(Hb1, gate_w3, gate_b3, scores, E);

    // ---- routing + outputs ----
    k_balanced<<<(B * E + 255) / 256, 256, 0, stream>>>(scores, conf, balWs, outBal, B * E);
    k_route<<<1, 1, 0, stream>>>(balWs, logitsE, outLogits, outD);
}
